// CrossAttention_37769942401112
// MI455X (gfx1250) — compile-verified, matrix operand order checked by reading, not run on hardware
//
#include <hip/hip_runtime.h>
#include <hip/hip_bf16.h>

typedef __attribute__((ext_vector_type(16))) __bf16 v16bf;
typedef __attribute__((ext_vector_type(8)))  float  v8f;

union BFrag { unsigned int u[8]; uint4 q[2]; v16bf v; };

__device__ __forceinline__ unsigned short f2bf(float f) {
    unsigned int u = __float_as_uint(f);
    u += 0x7FFFu + ((u >> 16) & 1u);
    return (unsigned short)(u >> 16);
}

__device__ __forceinline__ unsigned int pack2bf(float lo, float hi) {
    unsigned int ul = __float_as_uint(lo); ul += 0x7FFFu + ((ul >> 16) & 1u);
    unsigned int uh = __float_as_uint(hi); uh += 0x7FFFu + ((uh >> 16) & 1u);
    return (ul >> 16) | (uh & 0xFFFF0000u);
}

__device__ __forceinline__ v8f wmma_bf16(const BFrag& a, const BFrag& b, v8f c) {
    return __builtin_amdgcn_wmma_f32_16x16x32_bf16(
        false, a.v, false, b.v, (short)0, c, false, false);
}

// one 16x32 bf16 fragment = two aligned b128 loads (K pairs are contiguous:
// regs 0..3 hold K = kc*32+g*8..+7, regs 4..7 hold K = kc*32+16+g*8..+7)
__device__ __forceinline__ void load_frag_g(BFrag& f, const unsigned short* rowbase,
                                            int kc, int g) {
    const uint4* p = (const uint4*)(rowbase + kc * 32 + g * 8);
    f.q[0] = p[0];
    f.q[1] = p[2];
}

// ---------------------------------------------------------------------------
__global__ void zero_u32_kernel(unsigned int* p, int n) {
    int i = blockIdx.x * blockDim.x + threadIdx.x;
    if (i < n) p[i] = 0u;
}

__global__ void cvt_bf16_kernel(const float* __restrict__ in,
                                unsigned int* __restrict__ out,
                                int npairs, float scale) {
    int i = blockIdx.x * blockDim.x + threadIdx.x;
    if (i < npairs) {
        float2 v = ((const float2*)in)[i];
        out[i] = pack2bf(v.x * scale, v.y * scale);
    }
}

// ---------------------------------------------------------------------------
// KV projection: kT[b][h][tok(96)][dd], vT[b][h][dd][tok(96)].
// grid (5 t-tiles, 64 n-tiles, 16 b), block 32.
__global__ void kvproj_kernel(const unsigned short* __restrict__ ctxbf,
                              const unsigned short* __restrict__ wkvbf,
                              unsigned short* __restrict__ kT,
                              unsigned short* __restrict__ vT) {
    const int tt = blockIdx.x, nt = blockIdx.y, b = blockIdx.z;
    const int lane = threadIdx.x, g = lane >> 4, m = lane & 15;

    int tok = tt * 16 + m;
    int tokc = tok < 77 ? tok : 76; // rows >=77 never stored
    const unsigned short* arow = ctxbf + ((size_t)b * 77 + tokc) * 768;
    const unsigned short* wr = wkvbf + (size_t)(nt * 16 + m) * 768;

    v8f acc = {};
    for (int kc = 0; kc < 24; ++kc) {
        BFrag a, bb;
        load_frag_g(a, arow, kc, g);
        load_frag_g(bb, wr, kc, g);
        acc = wmma_bf16(a, bb, acc);
    }

    const int d = nt * 16 + m;
    #pragma unroll
    for (int r = 0; r < 8; ++r) {
        int t = tt * 16 + r + 8 * g;
        if (t < 77) {
            if (d < 512) {
                int h = d >> 6, dd = d & 63;
                kT[(((size_t)(b * 8 + h)) * 96 + t) * 64 + dd] = f2bf(acc[r]);
            } else {
                int d2 = d - 512, h = d2 >> 6, dd = d2 & 63;
                vT[(((size_t)(b * 8 + h)) * 64 + dd) * 96 + t] = f2bf(acc[r]);
            }
        }
    }
}

// ---------------------------------------------------------------------------
// Fused Q-projection + 8-head cross-attention + out-projection + RMSNorm +
// residual for one (b, 16-row spatial tile). grid (256, 16), block 32.
__global__ void __launch_bounds__(32)
fused_kernel(const float* __restrict__ x,
             const unsigned short* __restrict__ wqbf,   // pre-scaled by 1/sqrt(dh)
             const unsigned short* __restrict__ kT,
             const unsigned short* __restrict__ vT,
             const unsigned short* __restrict__ woutbf,
             const float* __restrict__ bout,
             const float* __restrict__ rms_scale,
             float* __restrict__ out) {
    const int pt = blockIdx.x, b = blockIdx.y;
    const int lane = threadIdx.x, g = lane >> 4, m = lane & 15;

    __shared__ __align__(16) unsigned short Abuf[16 * 256]; // x^T tile  [p][c]
    __shared__ __align__(16) unsigned short Qbuf[16 * 512]; // Q tile    [p][d]
    __shared__ __align__(16) unsigned short Plds[16 * 96];  // probs     [p][tok]
    __shared__ __align__(16) unsigned short Obuf[16 * 512]; // attn out  [p][d]
    __shared__ __align__(16) float          fbuf[16 * 256]; // proj out  [p][c]

    // ---- stage x^T tile, f32 -> bf16 on the fly (x read exactly once) ----
    for (int cc = 0; cc < 8; ++cc) {
        int c = cc * 32 + lane;
        const float* src = x + ((size_t)(b * 256) + c) * 4096 + pt * 16;
        #pragma unroll
        for (int q4 = 0; q4 < 4; ++q4) {
            float4 v = ((const float4*)src)[q4];
            Abuf[(q4 * 4 + 0) * 256 + c] = f2bf(v.x);
            Abuf[(q4 * 4 + 1) * 256 + c] = f2bf(v.y);
            Abuf[(q4 * 4 + 2) * 256 + c] = f2bf(v.z);
            Abuf[(q4 * 4 + 3) * 256 + c] = f2bf(v.w);
        }
    }
    __syncthreads();

    // A fragments for all 8 K chunks stay in VGPRs (64 regs)
    BFrag A[8];
    #pragma unroll
    for (int kc = 0; kc < 8; ++kc)
        load_frag_g(A[kc], &Abuf[m * 256], kc, g);

    // ---- Q projection: Q[p][d] = sum_c x^T[p][c] * Wq_bf[d][c] -----------
    for (int nt = 0; nt < 32; ++nt) {
        const unsigned short* wr = wqbf + (size_t)(nt * 16 + m) * 256;
        if (nt < 31) __builtin_prefetch(wr + 256, 0, 1);
        v8f acc = {};
        #pragma unroll
        for (int kc = 0; kc < 8; ++kc) {
            BFrag bb;
            load_frag_g(bb, wr, kc, g);
            acc = wmma_bf16(A[kc], bb, acc);
        }
        #pragma unroll
        for (int r = 0; r < 8; ++r)
            Qbuf[(r + 8 * g) * 512 + nt * 16 + m] = f2bf(acc[r]);
    }
    __syncthreads();

    // ---- attention per head ---------------------------------------------
    for (int h = 0; h < 8; ++h) {
        const size_t bh = (size_t)b * 8 + h;

        BFrag qa[2];
        #pragma unroll
        for (int kc = 0; kc < 2; ++kc)
            load_frag_g(qa[kc], &Qbuf[m * 512 + h * 64], kc, g);

        float S[6][8];
        #pragma unroll
        for (int nt = 0; nt < 6; ++nt) {
            v8f c = {};
            const unsigned short* krow = kT + (bh * 96 + nt * 16 + m) * 64;
            #pragma unroll
            for (int kc = 0; kc < 2; ++kc) {
                BFrag bb;
                load_frag_g(bb, krow, kc, g);
                c = wmma_bf16(qa[kc], bb, c);
            }
            int col = nt * 16 + m;
            #pragma unroll
            for (int r = 0; r < 8; ++r)
                S[nt][r] = (col < 77) ? c[r] : -1e30f;
        }

        // softmax: row r+8g lives across the 16 lanes of group g
        #pragma unroll
        for (int r = 0; r < 8; ++r) {
            float mx = S[0][r];
            #pragma unroll
            for (int nt = 1; nt < 6; ++nt) mx = fmaxf(mx, S[nt][r]);
            #pragma unroll
            for (int sh = 1; sh < 16; sh <<= 1) mx = fmaxf(mx, __shfl_xor(mx, sh, 32));
            float sum = 0.f;
            #pragma unroll
            for (int nt = 0; nt < 6; ++nt) { float e = __expf(S[nt][r] - mx); S[nt][r] = e; sum += e; }
            #pragma unroll
            for (int sh = 1; sh < 16; sh <<= 1) sum += __shfl_xor(sum, sh, 32);
            float inv = 1.f / sum;
            #pragma unroll
            for (int nt = 0; nt < 6; ++nt) S[nt][r] *= inv;
        }

        __syncthreads(); // protect Plds against previous head's readers
        #pragma unroll
        for (int nt = 0; nt < 6; ++nt)
            #pragma unroll
            for (int r = 0; r < 8; ++r)
                Plds[(r + 8 * g) * 96 + nt * 16 + m] = f2bf(S[nt][r]);
        __syncthreads();

        // O = P x V
        #pragma unroll
        for (int dt = 0; dt < 4; ++dt) {
            v8f c = {};
            const unsigned short* vrow = vT + (bh * 64 + dt * 16 + m) * 96;
            #pragma unroll
            for (int kc = 0; kc < 3; ++kc) {
                BFrag pa, vb;
                load_frag_g(pa, &Plds[m * 96], kc, g);
                load_frag_g(vb, vrow, kc, g);
                c = wmma_bf16(pa, vb, c);
            }
            #pragma unroll
            for (int r = 0; r < 8; ++r)
                Obuf[(r + 8 * g) * 512 + h * 64 + dt * 16 + m] = f2bf(c[r]);
        }
    }
    __syncthreads();

    // ---- output projection + bias ---------------------------------------
    for (int nt = 0; nt < 16; ++nt) {
        const unsigned short* wr = woutbf + (size_t)(nt * 16 + m) * 512;
        if (nt < 15) __builtin_prefetch(wr + 512, 0, 1);
        v8f acc = {};
        #pragma unroll
        for (int kc = 0; kc < 16; ++kc) {
            BFrag a, bb;
            load_frag_g(a, &Obuf[m * 512], kc, g);
            load_frag_g(bb, wr, kc, g);
            acc = wmma_bf16(a, bb, acc);
        }
        float bias = bout[nt * 16 + m];
        #pragma unroll
        for (int r = 0; r < 8; ++r)
            fbuf[(r + 8 * g) * 256 + nt * 16 + m] = acc[r] + bias;
    }
    __syncthreads();

    // ---- RMSNorm over channels + residual -------------------------------
    float ss = 0.f;
    for (int cc = 0; cc < 128; ++cc) {
        float v = fbuf[m * 256 + g * 128 + cc];
        ss += v * v;
    }
    ss += __shfl_xor(ss, 16, 32);
    float inv = rsqrtf(ss * (1.f / 256.f) + 1e-6f);

    const float* xb = x + ((size_t)b * 256) * 4096 + (size_t)pt * 16;
    float* ob = out + ((size_t)b * 256) * 4096 + (size_t)pt * 16;
    for (int c0 = 0; c0 < 256; c0 += 2) {
        int c = c0 + g;
        float v = fbuf[m * 256 + c];
        ob[(size_t)c * 4096 + m] = v * inv * rms_scale[c] + xb[(size_t)c * 4096 + m];
    }
}

// ---------------------------------------------------------------------------
extern "C" void kernel_launch(void* const* d_in, const int* in_sizes, int n_in,
                              void* d_out, int out_size, void* d_ws, size_t ws_size,
                              hipStream_t stream) {
    const float* x    = (const float*)d_in[0];
    const float* ctx  = (const float*)d_in[1];
    const float* Wq   = (const float*)d_in[2];
    const float* Wkv  = (const float*)d_in[3];
    const float* Wout = (const float*)d_in[4];
    const float* bo   = (const float*)d_in[5];
    const float* rs   = (const float*)d_in[6];
    float* out = (float*)d_out;

    char* ws = (char*)d_ws;
    size_t off = 0;
    auto take = [&](size_t bytes) { char* p = ws + off; off += bytes; return p; };

    unsigned short* kT     = (unsigned short*)take((size_t)16 * 8 * 96 * 64 * 2);
    unsigned short* vT     = (unsigned short*)take((size_t)16 * 8 * 96 * 64 * 2);
    unsigned short* ctxbf  = (unsigned short*)take((size_t)16 * 77 * 768 * 2);
    unsigned short* wqbf   = (unsigned short*)take((size_t)512 * 256 * 2);
    unsigned short* wkvbf  = (unsigned short*)take((size_t)1024 * 768 * 2);
    unsigned short* woutbf = (unsigned short*)take((size_t)256 * 512 * 2);

    const float qscale = 0.125f; // 1/sqrt(64), folded into Wq

    auto cvt = [&](const float* in, unsigned short* dst, size_t nelem, float s) {
        int npairs = (int)(nelem / 2);
        cvt_bf16_kernel<<<(npairs + 255) / 256, 256, 0, stream>>>(in, (unsigned int*)dst, npairs, s);
    };
    cvt(ctx,  ctxbf,  (size_t)16 * 77 * 768, 1.f);
    cvt(Wq,   wqbf,   (size_t)512 * 256,     qscale);
    cvt(Wkv,  wkvbf,  (size_t)1024 * 768,    1.f);
    cvt(Wout, woutbf, (size_t)256 * 512,     1.f);

    // zero kT+vT (contiguous): 2 * 786432 shorts = 786432 uints
    int nz = 786432;
    zero_u32_kernel<<<(nz + 255) / 256, 256, 0, stream>>>((unsigned int*)kT, nz);

    kvproj_kernel<<<dim3(5, 64, 16), 32, 0, stream>>>(ctxbf, wkvbf, kT, vT);
    fused_kernel<<<dim3(256, 16), 32, 0, stream>>>(x, wqbf, kT, vT, woutbf, bo, rs, out);
}
